// SelfAttention_62088047230984
// MI455X (gfx1250) — matrix operand order repaired and checked by reading, not compiled, not run on hardware
//
#include <hip/hip_runtime.h>

// ---------------------------------------------------------------------------
// Self-attention, N=8192, d=256, fp32 in/out.
// Compute-bound on WMMA: 2x ~34 GFLOP attention GEMMs; Q/K/V (12 MB bf16)
// fit in the 192 MB L2, HBM traffic is negligible at 23.3 TB/s.
// Path: fp32 -> bf16, v_wmma_f32_16x16x32_bf16 with f32 accumulation,
// flash-attention (online softmax), LDS round-trip for the P D->A transpose.
// ---------------------------------------------------------------------------

typedef __attribute__((ext_vector_type(16))) __bf16 bf16x16;
typedef __attribute__((ext_vector_type(8)))  __bf16 bf16x8;
typedef __attribute__((ext_vector_type(8)))  float  f32x8;

#define N_TOK 8192
#define DIM   256
#define NE    (N_TOK * DIM)        // 2,097,152 elements
#define NW    (DIM * DIM)          // 65,536 elements per weight matrix

static __device__ __forceinline__ bf16x16 mk16(bf16x8 lo, bf16x8 hi) {
  bf16x16 r;
#pragma unroll
  for (int i = 0; i < 8; ++i) { r[i] = lo[i]; r[i + 8] = hi[i]; }
  return r;
}

static __device__ __forceinline__ f32x8 wmma_bf16(bf16x16 a, bf16x16 b, f32x8 c) {
  // (neg_a, A, neg_b, B, c_mod, C, reuse_a, reuse_b)
  return __builtin_amdgcn_wmma_f32_16x16x32_bf16(false, a, false, b, (short)0, c,
                                                 false, false);
}

static __device__ __forceinline__ f32x8 f32x8_zero() {
  f32x8 z;
#pragma unroll
  for (int i = 0; i < 8; ++i) z[i] = 0.0f;
  return z;
}

// ---------------------------------------------------------------------------
// prep: embeddings fp32 -> bf16 (row-major), weights fp32 -> bf16 untransposed
// (Wt[n][k] = W[n][k]) so WMMA B-fragment loads are contiguous.
// ---------------------------------------------------------------------------
__global__ void prep_kernel(const float* __restrict__ E,
                            const float* __restrict__ Wq,
                            const float* __restrict__ Wk,
                            const float* __restrict__ Wv,
                            __bf16* __restrict__ Eb,
                            __bf16* __restrict__ Wqt,
                            __bf16* __restrict__ Wkt,
                            __bf16* __restrict__ Wvt) {
  int i = blockIdx.x * blockDim.x + threadIdx.x;
  if (i < NE) {
    Eb[i] = (__bf16)E[i];
  } else {
    int j = i - NE;
    if (j < NW) {
      int dk = j >> 8;       // row of Wt
      int n  = j & 255;      // col of Wt
      Wqt[j] = (__bf16)Wq[j];
      Wkt[j] = (__bf16)Wk[j];
      Wvt[j] = (__bf16)Wv[j];
    }
  }
}

// ---------------------------------------------------------------------------
// QKV projections: one wave per 16x16 output tile, 8 WMMA k-steps over d=256.
//   mat 0 -> Q  (row-major, pre-scaled by 1/sqrt(d) = 0.0625)
//   mat 1 -> K  (row-major like Q, unscaled, so S=Q*K^T B-reads
//                are contiguous rows of K)
//   mat 2 -> Vt (stored transposed: Vt[j][i] = V[i][j])
// ---------------------------------------------------------------------------
__global__ void qkv_kernel(const __bf16* __restrict__ Eb,
                           const __bf16* __restrict__ Wqt,
                           const __bf16* __restrict__ Wkt,
                           const __bf16* __restrict__ Wvt,
                           __bf16* __restrict__ Qb,
                           __bf16* __restrict__ Ktb,
                           __bf16* __restrict__ Vb) {
  const int lane = threadIdx.x & 31;
  const int wid  = (blockIdx.x * blockDim.x + threadIdx.x) >> 5;
  const int tilesPerMat = (N_TOK / 16) * (DIM / 16);   // 8192
  const int mat = wid / tilesPerMat;
  if (mat >= 3) return;                                 // wave-uniform
  const int rem = wid % tilesPerMat;
  const int i0  = (rem / (DIM / 16)) * 16;              // output row tile
  const int n0  = (rem % (DIM / 16)) * 16;              // output col tile
  const __bf16* Wt = (mat == 0) ? Wqt : (mat == 1) ? Wkt : Wvt;
  const int lm = lane & 15, half = lane >> 4;

  f32x8 acc = f32x8_zero();
#pragma unroll
  for (int c = 0; c < 8; ++c) {
    const int d0 = c * 32;
    // A fragment: row i0+lm, k-chunks at d0+8*half and +16 (16B loads)
    const __bf16* ap = Eb + (size_t)(i0 + lm) * DIM + d0 + half * 8;
    bf16x16 a = mk16(*(const bf16x8*)ap, *(const bf16x8*)(ap + 16));
    // B fragment: lane holds W row n0+lm, k-chunks at d0+8*half and +16 (16B loads)
    const __bf16* bp = Wt + (size_t)(n0 + lm) * DIM + d0 + half * 8;
    bf16x16 b = mk16(*(const bf16x8*)bp, *(const bf16x8*)(bp + 16));
    acc = wmma_bf16(a, b, acc);
  }

  if (mat == 2) {
    // transpose-store: element r is row i0+r+8*half, col n0+lm
    // -> Vt[(n0+lm)][i0 + 8*half + r]: r consecutive => one 16B store
    bf16x8 pk;
#pragma unroll
    for (int r = 0; r < 8; ++r) pk[r] = (__bf16)acc[r];
    *(bf16x8*)(Vb + (size_t)(n0 + lm) * N_TOK + i0 + half * 8) = pk;
  } else {
    __bf16* O = (mat == 0) ? Qb : Ktb;
    const float sc = (mat == 0) ? 0.0625f : 1.0f;   // fold 1/sqrt(256) into Q
#pragma unroll
    for (int r = 0; r < 8; ++r)
      O[(size_t)(i0 + r + half * 8) * DIM + n0 + lm] = (__bf16)(acc[r] * sc);
  }
}

// ---------------------------------------------------------------------------
// Flash attention: one wave owns 16 query rows, streams keys in blocks of 32.
// S tile = 16x32 via two 16x16 WMMA chains (8 k-chunks each); online softmax
// with half-wave shfl reductions (C/D layout: row = r + 8*(lane>=16),
// col = lane&15); P transposed D->A layout through per-wave LDS tile;
// O += P*V via 16 WMMAs (V stored transposed, B-reads are rows of Vt).
// ---------------------------------------------------------------------------
#define ATT_WAVES 2
__global__ void __launch_bounds__(32 * ATT_WAVES)
attn_kernel(const __bf16* __restrict__ Qb,
            const __bf16* __restrict__ Ktb,
            const __bf16* __restrict__ Vb,
            float* __restrict__ Out) {
  __shared__ __align__(64) __bf16 ldsP[ATT_WAVES][16 * 32];
  const int lane = threadIdx.x & 31;
  const int w    = threadIdx.x >> 5;
  const int q0   = (blockIdx.x * ATT_WAVES + w) * 16;
  const int lm = lane & 15, half = lane >> 4;
  __bf16* myP = &ldsP[w][0];

  f32x8 o[16];
#pragma unroll
  for (int t = 0; t < 16; ++t) o[t] = f32x8_zero();
  float mrow[8], lrow[8];
#pragma unroll
  for (int r = 0; r < 8; ++r) { mrow[r] = -1e30f; lrow[r] = 0.0f; }

  for (int kb = 0; kb < N_TOK; kb += 32) {
    // ---- S = (Q/sqrt(d)) * K^T : two 16x16 n-halves, 8 k-chunks ----
    f32x8 s0 = f32x8_zero(), s1 = f32x8_zero();
#pragma unroll
    for (int c = 0; c < 8; ++c) {
      const int d0 = c * 32;
      const __bf16* qp = Qb + (size_t)(q0 + lm) * DIM + d0 + half * 8;
      bf16x16 a = mk16(*(const bf16x8*)qp, *(const bf16x8*)(qp + 16));
      const __bf16* kp = Ktb + (size_t)(kb + lm) * DIM + d0 + half * 8;
      s0 = wmma_bf16(a, mk16(*(const bf16x8*)kp, *(const bf16x8*)(kp + 16)), s0);
      s1 = wmma_bf16(a, mk16(*(const bf16x8*)(kp + 16 * DIM), *(const bf16x8*)(kp + 16 * DIM + 16)), s1);
    }

    // ---- online softmax over this 32-key block ----
#pragma unroll
    for (int r = 0; r < 8; ++r) {
      float mx = fmaxf(s0[r], s1[r]);
#pragma unroll
      for (int off = 1; off < 16; off <<= 1)
        mx = fmaxf(mx, __shfl_xor(mx, off, 32));   // reduce within 16-lane half
      const float mn    = fmaxf(mrow[r], mx);
      const float alpha = __expf(mrow[r] - mn);
      const float p0    = __expf(s0[r] - mn);
      const float p1    = __expf(s1[r] - mn);
      float rs = p0 + p1;
#pragma unroll
      for (int off = 1; off < 16; off <<= 1)
        rs += __shfl_xor(rs, off, 32);
      lrow[r] = lrow[r] * alpha + rs;
      mrow[r] = mn;
#pragma unroll
      for (int t = 0; t < 16; ++t) o[t][r] *= alpha;
      // stage P (row-major 16x32) in LDS: row = r + 8*half, cols lm / lm+16
      const int row = r + half * 8;
      myP[row * 32 + lm]      = (__bf16)p0;
      myP[row * 32 + 16 + lm] = (__bf16)p1;
    }

    // same-wave DS ordering guarantee + compiler barrier
    asm volatile("s_wait_dscnt 0x0" ::: "memory");

    // ---- read P back in A-fragment layout: row lm, k-chunks 8*half, +16 ----
    const __bf16* pp = myP + lm * 32 + half * 8;
    bf16x16 pa = mk16(*(const bf16x8*)pp, *(const bf16x8*)(pp + 16));

    // ---- O += P * V : Vt row t*16+lm, key chunks kb+8*half and +16 ----
#pragma unroll
    for (int t = 0; t < 16; ++t) {
      const __bf16* vp = Vb + (size_t)(t * 16 + lm) * N_TOK + kb + half * 8;
      o[t] = wmma_bf16(pa, mk16(*(const bf16x8*)vp, *(const bf16x8*)(vp + 16)), o[t]);
    }
  }

  // ---- epilogue: normalize by row sums, store fp32 ----
  float inv[8];
#pragma unroll
  for (int r = 0; r < 8; ++r) inv[r] = (lrow[r] > 0.0f) ? (1.0f / lrow[r]) : 0.0f;
#pragma unroll
  for (int t = 0; t < 16; ++t)
#pragma unroll
    for (int r = 0; r < 8; ++r)
      Out[(size_t)(q0 + r + half * 8) * DIM + t * 16 + lm] = o[t][r] * inv[r];
}

// ---------------------------------------------------------------------------
extern "C" void kernel_launch(void* const* d_in, const int* in_sizes, int n_in,
                              void* d_out, int out_size, void* d_ws, size_t ws_size,
                              hipStream_t stream) {
  const float* E  = (const float*)d_in[0];
  const float* Wq = (const float*)d_in[1];
  const float* Wk = (const float*)d_in[2];
  const float* Wv = (const float*)d_in[3];
  float* Out = (float*)d_out;

  // workspace carve-out (256B aligned): ~17 MB total
  char* ws = (char*)d_ws;
  size_t off = 0;
  auto carve = [&](size_t bytes) -> char* {
    char* p = ws + off;
    off = (off + bytes + 255) & ~(size_t)255;
    return p;
  };
  __bf16* Eb  = (__bf16*)carve((size_t)NE * 2);
  __bf16* Qb  = (__bf16*)carve((size_t)NE * 2);
  __bf16* Ktb = (__bf16*)carve((size_t)NE * 2);
  __bf16* Vb  = (__bf16*)carve((size_t)NE * 2);
  __bf16* Wqt = (__bf16*)carve((size_t)NW * 2);
  __bf16* Wkt = (__bf16*)carve((size_t)NW * 2);
  __bf16* Wvt = (__bf16*)carve((size_t)NW * 2);

  // 1) fp32 -> bf16 casts (no transposes)
  {
    const int total = NE + NW;
    const int blk = 256;
    prep_kernel<<<(total + blk - 1) / blk, blk, 0, stream>>>(
        E, Wq, Wk, Wv, Eb, Wqt, Wkt, Wvt);
  }
  // 2) Q (pre-scaled), K, V^T via WMMA: 3 * 8192 tiles, 8 waves/block
  {
    const int waves  = 3 * (N_TOK / 16) * (DIM / 16);
    const int blk    = 256;                 // 8 waves
    const int blocks = waves / (blk / 32);  // 3072
    qkv_kernel<<<blocks, blk, 0, stream>>>(Eb, Wqt, Wkt, Wvt, Qb, Ktb, Vb);
  }
  // 3) flash attention: 512 waves (16 query rows each), 2 waves/block
  {
    const int blocks = (N_TOK / 16) / ATT_WAVES;  // 256
    attn_kernel<<<blocks, 32 * ATT_WAVES, 0, stream>>>(Qb, Ktb, Vb, Out);
  }
}
